// GatedDirGCNConv_37752762532077
// MI455X (gfx1250) — matrix operand order repaired and checked by reading, not compiled, not run on hardware
//
#include <hip/hip_runtime.h>
#include <hip/hip_bf16.h>

typedef __attribute__((ext_vector_type(16))) _Float16 v16h;
typedef __attribute__((ext_vector_type(8)))  _Float16 h8;
typedef __attribute__((ext_vector_type(8)))  float    v8f;
typedef __attribute__((ext_vector_type(4)))  int      v4i;

#define ALPHA_C 0.5f

__device__ __forceinline__ v8f wmma_f16(v16h a, v16h b, v8f c) {
  // 8 args: (neg_a, A, neg_b, B, c_mod, C, reuse_a, reuse_b)
  return __builtin_amdgcn_wmma_f32_16x16x32_f16(
      false, a, false, b, (short)0, c, false, false);
}

// ---- CDNA5 async global->LDS copy (ASYNCcnt tracked), with safe fallback
__device__ __forceinline__ void lds_async_copy_b128(_Float16* lds_dst,
                                                    const _Float16* g_src) {
#if __has_builtin(__builtin_amdgcn_global_load_async_to_lds_b128)
  __builtin_amdgcn_global_load_async_to_lds_b128(
      (v4i __attribute__((address_space(1)))*)g_src,
      (v4i __attribute__((address_space(3)))*)lds_dst, 0, 0);
#else
  *(h8*)lds_dst = *(const h8*)g_src;
#endif
}

__device__ __forceinline__ void lds_async_wait() {
#if __has_builtin(__builtin_amdgcn_s_wait_asynccnt)
  __builtin_amdgcn_s_wait_asynccnt(0);
#else
  asm volatile("s_wait_asynccnt 0x0" ::: "memory");
#endif
}

// ---------------------------------------------------------------- utilities
__global__ void zero_kernel(float* __restrict__ p, int n) {
  int i = blockIdx.x * blockDim.x + threadIdx.x;
  if (i < n) p[i] = 0.f;
}

__global__ void convert_f32_to_f16_kernel(const float* __restrict__ src,
                                          _Float16* __restrict__ dst, int n) {
  int i = blockIdx.x * blockDim.x + threadIdx.x;
  if (i < n) dst[i] = (_Float16)src[i];
}

// Prepack a row-major [Ktot x 64] f32 weight matrix into per-lane WMMA
// B-fragment order (f16). Fragment (nb,kb) holds cols nb*16..+15, K rows
// kb*32..+31.  Element t -> frag blk = t>>9, lane = (t>>4)&31, j = t&15.
// Lane l: col = nb*16+(l&15), K = kb*32 + (j>>3)*16 + (l>>4)*8 + (j&7).
__global__ void pack_weights_kernel(const float* __restrict__ W,
                                    _Float16* __restrict__ dst, int KB) {
  int t = blockIdx.x * blockDim.x + threadIdx.x;
  int total = KB * 4 * 512;
  if (t >= total) return;
  int j    = t & 15;
  int lane = (t >> 4) & 31;
  int blk  = t >> 9;
  int kb   = blk % KB;
  int nb   = blk / KB;
  int col  = nb * 16 + (lane & 15);
  int K    = kb * 32 + (j >> 3) * 16 + (lane >> 4) * 8 + (j & 7);
  dst[t] = (_Float16)W[K * 64 + col];
}

__global__ void degree_kernel(const float* __restrict__ counts,
                              const int* __restrict__ src,
                              const int* __restrict__ dst,
                              float* __restrict__ out_deg,
                              float* __restrict__ in_deg, int E) {
  int e = blockIdx.x * blockDim.x + threadIdx.x;
  if (e >= E) return;
  float c = counts[e];
  atomicAdd(&out_deg[src[e]], c);
  atomicAdd(&in_deg[dst[e]], c);
}

// ------------------------------------------------ h = x @ W + b  (WMMA f16)
__global__ void node_gemm_kernel(const _Float16* __restrict__ xh,
                                 const _Float16* __restrict__ wp,   // packed 4096
                                 const float* __restrict__ bias,
                                 float* __restrict__ hout, int N) {
  __shared__ __align__(32) _Float16 sW[4096];
  for (int t = threadIdx.x; t < 512; t += blockDim.x)
    lds_async_copy_b128(&sW[t * 8], wp + t * 8);
  lds_async_wait();
  __syncthreads();
  const int lane = threadIdx.x & 31;
  const int wid  = threadIdx.x >> 5;
  const int r = lane & 15, h = lane >> 4;
  const int wpb = blockDim.x >> 5;
  const int ntiles = N >> 4;
  for (int tile = blockIdx.x * wpb + wid; tile < ntiles; tile += gridDim.x * wpb) {
    const int n0 = tile << 4;
    const int node = n0 + r;
    v16h a[2];
#pragma unroll
    for (int kb = 0; kb < 2; ++kb) {
      const _Float16* base = xh + node * 64 + kb * 32 + h * 8;
      union { v16h v; h8 p[2]; } u;
      u.p[0] = *(const h8*)(base);
      u.p[1] = *(const h8*)(base + 16);
      a[kb] = u.v;
    }
    v8f acc[4] = {};   // first WMMA in each chain uses inline-constant C=0
#pragma unroll
    for (int nb = 0; nb < 4; ++nb)
#pragma unroll
      for (int kb = 0; kb < 2; ++kb) {
        v16h b = *((const v16h*)sW + (nb * 2 + kb) * 32 + lane);
        acc[nb] = wmma_f16(a[kb], b, acc[nb]);
      }
#pragma unroll
    for (int nb = 0; nb < 4; ++nb) {
      const float bv = bias[nb * 16 + r];
#pragma unroll
      for (int vr = 0; vr < 8; ++vr)
        hout[(n0 + vr + 8 * h) * 64 + nb * 16 + r] = acc[nb][vr] + bv;
    }
  }
}

// -------------------------------------- per-edge LCS MLP + weighted scatter
__global__ void edge_kernel(const _Float16* __restrict__ xh,
                            const float* __restrict__ counts,
                            const int* __restrict__ src,
                            const int* __restrict__ dst,
                            const _Float16* __restrict__ w1p,   // packed 8192
                            const float* __restrict__ b_l1,
                            const float* __restrict__ W_l2,
                            const float* __restrict__ b_l2,
                            const float* __restrict__ in_deg,
                            const float* __restrict__ out_deg,
                            const float* __restrict__ h_s2d,
                            const float* __restrict__ h_d2s,
                            float* __restrict__ m_in,
                            float* __restrict__ m_out, int E) {
  __shared__ __align__(32) _Float16 sW1[8192];
  __shared__ float sB1[64];
  __shared__ float sW2[64];
  __shared__ float szb[8][16];
  for (int t = threadIdx.x; t < 1024; t += blockDim.x)
    lds_async_copy_b128(&sW1[t * 8], w1p + t * 8);
  lds_async_wait();
  if (threadIdx.x < 64) {
    sB1[threadIdx.x] = b_l1[threadIdx.x];
    sW2[threadIdx.x] = W_l2[threadIdx.x];
  }
  __syncthreads();
  const float bl2 = b_l2[0];
  const int lane = threadIdx.x & 31;
  const int wid  = threadIdx.x >> 5;
  const int r = lane & 15, h = lane >> 4;
  const int wpb = blockDim.x >> 5;
  const int ntiles = E >> 4;
  volatile float* zb = szb[wid];

  for (int tile = blockIdx.x * wpb + wid; tile < ntiles; tile += gridDim.x * wpb) {
    const int e = (tile << 4) + r;
    const int s = src[e];
    const int d = dst[e];
    // prefetch the streamed src/dst index arrays ahead (global_prefetch_b8)
    __builtin_prefetch(src + e + 2048, 0, 0);
    __builtin_prefetch(dst + e + 2048, 0, 0);
    // A = [16 edges x 128] of concat(x_src, x_dst) in f16
    v16h a[4];
#pragma unroll
    for (int kb = 0; kb < 4; ++kb) {
      const int node = (kb < 2) ? s : d;
      const _Float16* base = xh + node * 64 + ((kb & 1) << 5) + h * 8;
      union { v16h v; h8 p[2]; } u;
      u.p[0] = *(const h8*)(base);
      u.p[1] = *(const h8*)(base + 16);
      a[kb] = u.v;
    }
    v8f acc[4] = {};   // inline-0 C on first WMMA of each chain
#pragma unroll
    for (int nb = 0; nb < 4; ++nb)
#pragma unroll
      for (int kb = 0; kb < 4; ++kb) {
        v16h b = *((const v16h*)sW1 + (nb * 4 + kb) * 32 + lane);
        acc[nb] = wmma_f16(a[kb], b, acc[nb]);
      }
    // second layer: z[M] = b_l2 + sum_col relu(hid[M,col] + b_l1[col]) * W_l2[col]
    float part[8] = {0.f, 0.f, 0.f, 0.f, 0.f, 0.f, 0.f, 0.f};
#pragma unroll
    for (int nb = 0; nb < 4; ++nb) {
      const float w2 = sW2[nb * 16 + r];
      const float bb = sB1[nb * 16 + r];
#pragma unroll
      for (int vr = 0; vr < 8; ++vr)
        part[vr] += fmaxf(acc[nb][vr] + bb, 0.f) * w2;
    }
#pragma unroll
    for (int m = 1; m < 16; m <<= 1)
#pragma unroll
      for (int vr = 0; vr < 8; ++vr)
        part[vr] += __shfl_xor(part[vr], m, 16);
    if (r == 0) {
#pragma unroll
      for (int vr = 0; vr < 8; ++vr) zb[h * 8 + vr] = part[vr];
    }
    const float z = zb[r] + bl2;           // sigmoid(z)>=0.5  <=>  z>=0
    const float keep = (z >= 0.f) ? 1.f : 0.f;
    const float od = fmaxf(out_deg[s], 1.f);
    const float idg = fmaxf(in_deg[d], 1.f);
    const float w_e = counts[e] * keep * rsqrtf(od * idg);
    if (w_e != 0.f) {
      const float* hs = h_s2d + s * 64 + h * 32;
      const float* hd = h_d2s + d * 64 + h * 32;
      float* mi = m_in + d * 64 + h * 32;
      float* mo = m_out + s * 64 + h * 32;
#pragma unroll
      for (int q = 0; q < 8; ++q) {
        float4 vs = *(const float4*)(hs + 4 * q);
        float4 vd = *(const float4*)(hd + 4 * q);
        atomicAdd(mi + 4 * q + 0, w_e * vs.x);
        atomicAdd(mi + 4 * q + 1, w_e * vs.y);
        atomicAdd(mi + 4 * q + 2, w_e * vs.z);
        atomicAdd(mi + 4 * q + 3, w_e * vs.w);
        atomicAdd(mo + 4 * q + 0, w_e * vd.x);
        atomicAdd(mo + 4 * q + 1, w_e * vd.y);
        atomicAdd(mo + 4 * q + 2, w_e * vd.z);
        atomicAdd(mo + 4 * q + 3, w_e * vd.w);
      }
    }
  }
}

// ------------------------------- gate MLP (WMMA) + alpha blend + residual
__global__ void gate_out_kernel(const float* __restrict__ m_in,
                                const float* __restrict__ m_out,
                                const float* __restrict__ x,
                                const _Float16* __restrict__ wg1p,  // packed 8192
                                const float* __restrict__ b_g1,
                                const float* __restrict__ W_g2,
                                const float* __restrict__ b_g2,
                                float* __restrict__ out, int N) {
  __shared__ __align__(32) _Float16 sW1[8192];
  __shared__ float sB1[64];
  __shared__ float sW2[64];
  __shared__ float szb[8][16];
  for (int t = threadIdx.x; t < 1024; t += blockDim.x)
    lds_async_copy_b128(&sW1[t * 8], wg1p + t * 8);
  lds_async_wait();
  if (threadIdx.x < 64) {
    sB1[threadIdx.x] = b_g1[threadIdx.x];
    sW2[threadIdx.x] = W_g2[threadIdx.x];
  }
  __syncthreads();
  const float bg2 = b_g2[0];
  const int lane = threadIdx.x & 31;
  const int wid  = threadIdx.x >> 5;
  const int r = lane & 15, h = lane >> 4;
  const int wpb = blockDim.x >> 5;
  const int ntiles = N >> 4;
  volatile float* zb = szb[wid];

  for (int tile = blockIdx.x * wpb + wid; tile < ntiles; tile += gridDim.x * wpb) {
    const int n0 = tile << 4;
    const int node = n0 + r;
    v16h a[4];
#pragma unroll
    for (int kb = 0; kb < 4; ++kb) {
      const float* srcp = (kb < 2) ? m_in : m_out;
      const float* base = srcp + node * 64 + ((kb & 1) << 5) + h * 8;
      v16h av;
#pragma unroll
      for (int j = 0; j < 8; ++j) {
        av[j]     = (_Float16)base[j];
        av[8 + j] = (_Float16)base[16 + j];
      }
      a[kb] = av;
    }
    v8f acc[4] = {};
#pragma unroll
    for (int nb = 0; nb < 4; ++nb)
#pragma unroll
      for (int kb = 0; kb < 4; ++kb) {
        v16h b = *((const v16h*)sW1 + (nb * 4 + kb) * 32 + lane);
        acc[nb] = wmma_f16(a[kb], b, acc[nb]);
      }
    float part[8] = {0.f, 0.f, 0.f, 0.f, 0.f, 0.f, 0.f, 0.f};
#pragma unroll
    for (int nb = 0; nb < 4; ++nb) {
      const float w2 = sW2[nb * 16 + r];
      const float bb = sB1[nb * 16 + r];
#pragma unroll
      for (int vr = 0; vr < 8; ++vr)
        part[vr] += fmaxf(acc[nb][vr] + bb, 0.f) * w2;
    }
#pragma unroll
    for (int m = 1; m < 16; m <<= 1)
#pragma unroll
      for (int vr = 0; vr < 8; ++vr)
        part[vr] += __shfl_xor(part[vr], m, 16);
    if (r == 0) {
#pragma unroll
      for (int vr = 0; vr < 8; ++vr) zb[h * 8 + vr] = part[vr];
    }
    const float z = zb[r] + bg2;
    const float g = 1.f / (1.f + __expf(-z));
    const float ga = ALPHA_C * g;
    const float gb = (1.f - ALPHA_C) * (1.f - g);
    const float* mi = m_in + node * 64 + h * 32;
    const float* mo = m_out + node * 64 + h * 32;
    const float* xx = x + node * 64 + h * 32;
    float* op = out + node * 64 + h * 32;
#pragma unroll
    for (int q = 0; q < 8; ++q) {
      float4 a4 = *(const float4*)(mi + 4 * q);
      float4 b4 = *(const float4*)(mo + 4 * q);
      float4 x4 = *(const float4*)(xx + 4 * q);
      float4 o4;
      o4.x = ga * a4.x + gb * b4.x + x4.x;
      o4.y = ga * a4.y + gb * b4.y + x4.y;
      o4.z = ga * a4.z + gb * b4.z + x4.z;
      o4.w = ga * a4.w + gb * b4.w + x4.w;
      *(float4*)(op + 4 * q) = o4;
    }
  }
}

// ---------------------------------------------------------------- launcher
extern "C" void kernel_launch(void* const* d_in, const int* in_sizes, int n_in,
                              void* d_out, int out_size, void* d_ws, size_t ws_size,
                              hipStream_t stream) {
  const float* x      = (const float*)d_in[0];
  const float* counts = (const float*)d_in[1];
  const float* W_s2d  = (const float*)d_in[2];
  const float* b_s2d  = (const float*)d_in[3];
  const float* W_d2s  = (const float*)d_in[4];
  const float* b_d2s  = (const float*)d_in[5];
  const float* W_l1   = (const float*)d_in[6];
  const float* b_l1   = (const float*)d_in[7];
  const float* W_l2   = (const float*)d_in[8];
  const float* b_l2   = (const float*)d_in[9];
  const float* W_g1   = (const float*)d_in[10];
  const float* b_g1   = (const float*)d_in[11];
  const float* W_g2   = (const float*)d_in[12];
  const float* b_g2   = (const float*)d_in[13];
  const int*   src    = (const int*)d_in[14];
  const int*   dst    = (const int*)d_in[15];

  const int N = in_sizes[0] / 64;
  const int E = in_sizes[14];

  // workspace carve-out (256B-aligned chunks)
  char* w = (char*)d_ws;
  auto carve = [&](size_t bytes) -> char* {
    char* p = w;
    w += (bytes + 255) & ~(size_t)255;
    return p;
  };
  // zeroed accumulators, contiguous so one zero_kernel covers everything
  const int nzero = 2 * N * 64 + 2 * N;
  float* m_in    = (float*)carve((size_t)nzero * 4);
  float* m_out   = m_in + (size_t)N * 64;
  float* in_deg  = m_out + (size_t)N * 64;
  float* out_deg = in_deg + N;

  _Float16* xh     = (_Float16*)carve((size_t)N * 64 * 2);
  float*    h_s2d  = (float*)carve((size_t)N * 64 * 4);
  float*    h_d2s  = (float*)carve((size_t)N * 64 * 4);
  _Float16* w1p    = (_Float16*)carve(8192 * 2);
  _Float16* ws2dp  = (_Float16*)carve(4096 * 2);
  _Float16* wd2sp  = (_Float16*)carve(4096 * 2);
  _Float16* wg1p   = (_Float16*)carve(8192 * 2);

  const int T = 256;

  zero_kernel<<<(nzero + T - 1) / T, T, 0, stream>>>(m_in, nzero);
  convert_f32_to_f16_kernel<<<(N * 64 + T - 1) / T, T, 0, stream>>>(x, xh, N * 64);
  pack_weights_kernel<<<(4 * 4 * 512 + T - 1) / T, T, 0, stream>>>(W_l1, w1p, 4);
  pack_weights_kernel<<<(2 * 4 * 512 + T - 1) / T, T, 0, stream>>>(W_s2d, ws2dp, 2);
  pack_weights_kernel<<<(2 * 4 * 512 + T - 1) / T, T, 0, stream>>>(W_d2s, wd2sp, 2);
  pack_weights_kernel<<<(4 * 4 * 512 + T - 1) / T, T, 0, stream>>>(W_g1, wg1p, 4);
  degree_kernel<<<(E + T - 1) / T, T, 0, stream>>>(counts, src, dst, out_deg, in_deg, E);

  const int ntilesN = N >> 4;                 // N divisible by 16 (50000)
  const int ntilesE = E >> 4;                 // E divisible by 16 (800000)
  const int wavesPerBlock = T / 32;
  const int gemmBlocks = (ntilesN + wavesPerBlock - 1) / wavesPerBlock;
  const int edgeBlocks = (ntilesE + wavesPerBlock - 1) / wavesPerBlock;

  node_gemm_kernel<<<gemmBlocks, T, 0, stream>>>(xh, ws2dp, b_s2d, h_s2d, N);
  node_gemm_kernel<<<gemmBlocks, T, 0, stream>>>(xh, wd2sp, b_d2s, h_d2s, N);

  edge_kernel<<<edgeBlocks, T, 0, stream>>>(xh, counts, src, dst, w1p, b_l1,
                                            W_l2, b_l2, in_deg, out_deg,
                                            h_s2d, h_d2s, m_in, m_out, E);

  gate_out_kernel<<<gemmBlocks, T, 0, stream>>>(m_in, m_out, x, wg1p, b_g1,
                                                W_g2, b_g2, (float*)d_out, N);
}
